// Stochastic_Duration_Predictor_20169166422240
// MI455X (gfx1250) — matrix operand order repaired and checked by reading, not compiled, not run on hardware
//
#include <hip/hip_runtime.h>
#include <hip/hip_bf16.h>
#include <math.h>

// ---------------------------------------------------------------------------
// Types for CDNA5 WMMA
// ---------------------------------------------------------------------------
typedef __bf16 v16bf __attribute__((ext_vector_type(16)));
typedef float  v8f   __attribute__((ext_vector_type(8)));

#define BATCH 32
#define TLEN  2048
#define CCALC 256
#define CIN   192
#define LOG2PI 1.8378770664093453f

__device__ __forceinline__ unsigned bf16bits(float f) {
    unsigned u = __float_as_uint(f);
    unsigned r = u + 0x7FFFu + ((u >> 16) & 1u);   // round-to-nearest-even
    return r >> 16;
}
__device__ __forceinline__ float softplusf(float x) {
    return (x > 20.f) ? x : log1pf(expf(x));
}

// ---------------------------------------------------------------------------
// Weight pack: f32 [M][K] row-major  ->  bf16 A-fragments
// layout: [mtile][kstep][lane(32)][j(16)]   (v16bf per lane per fragment)
//   j<8 : K = kstep*32 + (lane>>4)*8 + j
//   j>=8: K = kstep*32 + 16 + (lane>>4)*8 + (j-8)
//   M   = mtile*16 + (lane&15)
// ---------------------------------------------------------------------------
__global__ void pack_weights_k(const float* __restrict__ W,
                               unsigned short* __restrict__ out,
                               int M_real, int K, int Mtiles) {
    int Ksteps = K / 32;
    long long total = (long long)Mtiles * Ksteps * 512;
    long long idx = (long long)blockIdx.x * 256 + threadIdx.x;
    if (idx >= total) return;
    int j    = (int)(idx & 15);
    int lane = (int)((idx >> 4) & 31);
    int ks   = (int)((idx >> 9) % Ksteps);
    int mt   = (int)(idx / ((long long)Ksteps * 512));
    int m  = mt * 16 + (lane & 15);
    int kg = lane >> 4;
    int k  = ks * 32 + ((j < 8) ? (kg * 8 + j) : (16 + kg * 8 + (j - 8)));
    float v = (m < M_real) ? W[(size_t)m * K + k] : 0.f;
    out[idx] = (unsigned short)bf16bits(v);
}

// ---------------------------------------------------------------------------
// GEMM:  Y[b][m][t] = sum_k A[m][k] * X[b][k][t]  (+ bias[m])
// bf16 WMMA 16x16x32, f32 accumulate.  Block: 256 thr = 8 waves.
// Block tile: M = Mtiles*16 (waves own 2 m-tiles each), N = 64.
// Optional mask: X column t multiplied by mask[b][t] when staged.
// ---------------------------------------------------------------------------
__global__ __launch_bounds__(256)
void gemm_wmma_k(const float* __restrict__ Xin,
                 float* __restrict__ Yout,
                 const unsigned short* __restrict__ Apack,
                 const float* __restrict__ bias,
                 const float* __restrict__ maskBT,
                 int K, int Mtiles, int M_real, int Tlen) {
    const int b    = blockIdx.y;
    const int tn0  = blockIdx.x * 64;
    const int tid  = threadIdx.x;
    const int lane = tid & 31;
    const int wave = tid >> 5;
    const int Ksteps = K / 32;

    __shared__ __attribute__((aligned(32))) unsigned short ldsB[4 * 32 * 16];
    unsigned int* ldsB32 = (unsigned int*)ldsB;

    const float* Xb = Xin + (size_t)b * K * Tlen;

    // staging mapping: 256 threads -> 16 k-pairs x 16 float4 columns
    const int kp = tid >> 4;          // 0..15 -> k0 = kp*2
    const int n4 = (tid & 15) << 2;   // 0,4,...,60
    const int k0 = kp << 1;
    const int grp = ((k0 >> 3) & 1) << 4;   // 0 or 16 (lane group by K half)
    const int jd  = ((k0 & 7) >> 1) + ((k0 >> 4) << 2);   // dword index 0..7

    float mc[4];
    if (maskBT) {
        const float* mb = maskBT + (size_t)b * Tlen + tn0 + n4;
        mc[0] = mb[0]; mc[1] = mb[1]; mc[2] = mb[2]; mc[3] = mb[3];
    } else {
        mc[0] = mc[1] = mc[2] = mc[3] = 1.f;
    }

    v8f acc[2][4];
#pragma unroll
    for (int i = 0; i < 2; ++i)
#pragma unroll
        for (int j = 0; j < 4; ++j)
            acc[i][j] = (v8f){0.f, 0.f, 0.f, 0.f, 0.f, 0.f, 0.f, 0.f};

    const int  mt0 = wave * 2;
    const bool doW = (mt0 < Mtiles);

    for (int ks = 0; ks < Ksteps; ++ks) {
        // ---- stage B tile (32 x 64) into LDS in fragment order ----
        const float* r0 = Xb + (size_t)(ks * 32 + k0) * Tlen + tn0 + n4;
        const float4 x0 = *(const float4*)r0;
        const float4 x1 = *(const float4*)(r0 + Tlen);
        float a0v[4] = {x0.x, x0.y, x0.z, x0.w};
        float a1v[4] = {x1.x, x1.y, x1.z, x1.w};
#pragma unroll
        for (int e = 0; e < 4; ++e) {
            int n = n4 + e;
            int ntile = n >> 4;
            int laneW = grp + (n & 15);
            unsigned lo = bf16bits(a0v[e] * mc[e]);
            unsigned hi = bf16bits(a1v[e] * mc[e]);
            ldsB32[(ntile * 32 + laneW) * 8 + jd] = lo | (hi << 16);
        }
        __syncthreads();

        if (doW) {
            v16bf a0 = *reinterpret_cast<const v16bf*>(
                Apack + ((size_t)(mt0 * Ksteps + ks) * 32 + lane) * 16);
            v16bf a1 = *reinterpret_cast<const v16bf*>(
                Apack + ((size_t)((mt0 + 1) * Ksteps + ks) * 32 + lane) * 16);
#pragma unroll
            for (int nt = 0; nt < 4; ++nt) {
                v16bf bf = *reinterpret_cast<const v16bf*>(&ldsB[(nt * 32 + lane) * 16]);
                acc[0][nt] = __builtin_amdgcn_wmma_f32_16x16x32_bf16(
                    false, a0, false, bf, (short)0, acc[0][nt], false, false);
                acc[1][nt] = __builtin_amdgcn_wmma_f32_16x16x32_bf16(
                    false, a1, false, bf, (short)0, acc[1][nt], false, false);
            }
        }
        __syncthreads();
    }

    if (doW) {
#pragma unroll
        for (int mt = 0; mt < 2; ++mt) {
            int mbase = (mt0 + mt) * 16 + ((lane >> 4) << 3);
#pragma unroll
            for (int nt = 0; nt < 4; ++nt) {
                int col = tn0 + nt * 16 + (lane & 15);
#pragma unroll
                for (int r = 0; r < 8; ++r) {
                    int row = mbase + r;
                    float bv = (row < M_real) ? bias[row] : 0.f;
                    Yout[((size_t)b * Mtiles * 16 + row) * Tlen + col] =
                        acc[mt][nt][r] + bv;
                }
            }
        }
    }
}

// ---------------------------------------------------------------------------
// Depthwise conv k=3 dilated, input pre-multiplied by mask, + bias.
// ---------------------------------------------------------------------------
__global__ __launch_bounds__(256)
void dwconv_k(const float* __restrict__ X, float* __restrict__ Y,
              const float* __restrict__ Wdw, const float* __restrict__ Bdw,
              const float* __restrict__ maskBT, int dil, int Tlen, int C) {
    int b = blockIdx.y;
    int t = blockIdx.x * 64 + (threadIdx.x & 63);
    int cbase = threadIdx.x >> 6;
    const float* mb = maskBT + (size_t)b * Tlen;
    int tm = t - dil, tp = t + dil;
    float m0 = mb[t];
    float mm = (tm >= 0) ? mb[tm] : 0.f;
    float mp = (tp < Tlen) ? mb[tp] : 0.f;
    for (int ci = 0; ci < C / 4; ++ci) {
        int c = ci * 4 + cbase;
        const float* xr = X + ((size_t)b * C + c) * Tlen;
        float w0 = Wdw[c * 3 + 0], w1 = Wdw[c * 3 + 1], w2 = Wdw[c * 3 + 2];
        float vm = (tm >= 0)  ? xr[tm] * mm : 0.f;
        float vc = xr[t] * m0;
        float vp = (tp < Tlen) ? xr[tp] * mp : 0.f;
        Y[((size_t)b * C + c) * Tlen + t] = w0 * vm + w1 * vc + w2 * vp + Bdw[c];
    }
}

// ---------------------------------------------------------------------------
// LayerNorm over channels (per (b,t)) + exact GELU + optional residual add.
// ---------------------------------------------------------------------------
__global__ __launch_bounds__(256)
void ln_gelu_k(const float* __restrict__ X, float* __restrict__ Y,
               const float* __restrict__ G, const float* __restrict__ Bt,
               const float* __restrict__ Res, int Tlen, int C) {
    int b = blockIdx.y;
    int tt = threadIdx.x & 63;
    int part = threadIdx.x >> 6;     // 0..3
    int t = blockIdx.x * 64 + tt;
    __shared__ float s_sum[4][64], s_sq[4][64], s_mu[64], s_rs[64];
    const float* xb = X + (size_t)b * C * Tlen + t;
    int c0 = part * (C / 4);
    float sum = 0.f, sq = 0.f;
    for (int i = 0; i < C / 4; ++i) {
        float v = xb[(size_t)(c0 + i) * Tlen];
        sum += v; sq += v * v;
    }
    s_sum[part][tt] = sum; s_sq[part][tt] = sq;
    __syncthreads();
    if (part == 0) {
        float s = s_sum[0][tt] + s_sum[1][tt] + s_sum[2][tt] + s_sum[3][tt];
        float q = s_sq[0][tt] + s_sq[1][tt] + s_sq[2][tt] + s_sq[3][tt];
        float mu = s / (float)C;
        float var = q / (float)C - mu * mu;
        s_mu[tt] = mu;
        s_rs[tt] = rsqrtf(var + 1e-5f);
    }
    __syncthreads();
    float mu = s_mu[tt], rs = s_rs[tt];
    float* yb = Y + (size_t)b * C * Tlen + t;
    const float* rb = Res ? (Res + (size_t)b * C * Tlen + t) : (const float*)0;
    for (int i = 0; i < C / 4; ++i) {
        size_t off = (size_t)(c0 + i) * Tlen;
        float v = xb[off];
        float g = (v - mu) * rs * G[c0 + i] + Bt[c0 + i];
        float act = 0.5f * g * (1.f + erff(g * 0.70710678118654752f));
        yb[off] = rb ? (act + rb[off]) : act;
    }
}

// ---------------------------------------------------------------------------
// 1->C channel expand conv (+optional condition add): Y = w[c]*x0 + b[c] + cond
// ---------------------------------------------------------------------------
__global__ __launch_bounds__(256)
void ch_expand_k(const float* __restrict__ X0, int x0_bstride,
                 const float* __restrict__ Wc, const float* __restrict__ Bc,
                 const float* __restrict__ Cond, float* __restrict__ Y,
                 int Tlen, int C) {
    int b = blockIdx.y;
    int t = blockIdx.x * 64 + (threadIdx.x & 63);
    int cbase = threadIdx.x >> 6;
    float xv = X0[(size_t)b * x0_bstride + t];
    for (int ci = 0; ci < C / 4; ++ci) {
        int c = ci * 4 + cbase;
        size_t o = ((size_t)b * C + c) * Tlen + t;
        float v = Wc[c] * xv + Bc[c];
        if (Cond) v += Cond[o];
        Y[o] = v;
    }
}

// Out = (Xa?Xa:0) + Yg*mask   over [B][C][T]
__global__ void axpy_mask_k(const float* __restrict__ Xa, const float* __restrict__ Yg,
                            const float* __restrict__ maskBT, float* __restrict__ Out,
                            int Tlen, int C) {
    size_t idx = (size_t)blockIdx.x * 256 + threadIdx.x;
    int t = (int)(idx % Tlen);
    int b = (int)(idx / ((size_t)Tlen * C));
    float m = maskBT[(size_t)b * Tlen + t];
    float v = Yg[idx] * m;
    if (Xa) v += Xa[idx];
    Out[idx] = v;
}

__global__ void mask_k(const int* __restrict__ len, float* __restrict__ m, int Tlen) {
    int t = blockIdx.x * 256 + threadIdx.x;
    int b = blockIdx.y;
    m[(size_t)b * Tlen + t] = (t < len[b]) ? 1.f : 0.f;
}

__global__ void fill0_k(float* p, int n) {
    int i = blockIdx.x * 256 + threadIdx.x;
    if (i < n) p[i] = 0.f;
}

// elementwise affine over [B][2][T]:  Out = (m[ch] + exp(ls[ch])*X) * mask
__global__ void affine_k(const float* __restrict__ Xin, const float* __restrict__ m2,
                         const float* __restrict__ ls2, const float* __restrict__ mask,
                         float* __restrict__ Out, int Tlen) {
    int idx = blockIdx.x * 256 + threadIdx.x;
    int t = idx % Tlen;
    int ch = (idx / Tlen) & 1;
    int b = idx / (2 * Tlen);
    float mk = mask[(size_t)b * Tlen + t];
    Out[idx] = (m2[ch] + expf(ls2[ch]) * Xin[idx]) * mk;
}

// ---------------------------------------------------------------------------
// Rational-quadratic spline flow step (K=10 bins), acting on channel 1,
// then writes flipped output: ch0' = rqs(x1)*mk, ch1' = x0*mk. lad accumulates.
// ---------------------------------------------------------------------------
__global__ __launch_bounds__(256)
void rqs_k(const float* __restrict__ Proj, float* __restrict__ ZQ,
           const float* __restrict__ maskBT, float* __restrict__ LadAcc, int Tlen) {
    int idx = blockIdx.x * 256 + threadIdx.x;
    int t = idx % Tlen, b = idx / Tlen;
    const float tb = 5.f, minb = 0.001f;
    const float* hp = Proj + ((size_t)b * 32) * Tlen + t;
    size_t zo = (size_t)b * 2 * Tlen + t;
    float x0 = ZQ[zo], x1 = ZQ[zo + Tlen];
    float mk = maskBT[idx];

    float cumw[11], cumh[11], d[11];
    {
        float u[10];
#pragma unroll
        for (int k = 0; k < 10; ++k) u[k] = hp[(size_t)k * Tlen] * 0.0625f;
        float mx = u[0];
#pragma unroll
        for (int k = 1; k < 10; ++k) mx = fmaxf(mx, u[k]);
        float s = 0.f, e[10];
#pragma unroll
        for (int k = 0; k < 10; ++k) { e[k] = expf(u[k] - mx); s += e[k]; }
        float cs = 0.f; cumw[0] = -tb;
#pragma unroll
        for (int k = 0; k < 10; ++k) {
            float wk = minb + (1.f - minb * 10.f) * (e[k] / s);
            cs += wk;
            cumw[k + 1] = 2.f * tb * cs - tb;
        }
        cumw[10] = tb;
    }
    {
        float u[10];
#pragma unroll
        for (int k = 0; k < 10; ++k) u[k] = hp[(size_t)(10 + k) * Tlen] * 0.0625f;
        float mx = u[0];
#pragma unroll
        for (int k = 1; k < 10; ++k) mx = fmaxf(mx, u[k]);
        float s = 0.f, e[10];
#pragma unroll
        for (int k = 0; k < 10; ++k) { e[k] = expf(u[k] - mx); s += e[k]; }
        float cs = 0.f; cumh[0] = -tb;
#pragma unroll
        for (int k = 0; k < 10; ++k) {
            float hk = minb + (1.f - minb * 10.f) * (e[k] / s);
            cs += hk;
            cumh[k + 1] = 2.f * tb * cs - tb;
        }
        cumh[10] = tb;
    }
    d[0] = 1.f; d[10] = 1.f;   // softplus(log(exp(1-min_d)-1)) + min_d == 1
#pragma unroll
    for (int i = 1; i <= 9; ++i)
        d[i] = 0.001f + softplusf(hp[(size_t)(19 + i) * Tlen]);

    float x = fminf(fmaxf(x1, -tb), tb);
    int bi = 0;
#pragma unroll
    for (int i = 0; i < 11; ++i) bi += (x >= cumw[i]) ? 1 : 0;
    bi -= 1; bi = bi < 0 ? 0 : (bi > 9 ? 9 : bi);

    float icw = cumw[bi], iw = cumw[bi + 1] - cumw[bi];
    float ich = cumh[bi], ih = cumh[bi + 1] - cumh[bi];
    float delta = ih / iw;
    float d0 = d[bi], d1 = d[bi + 1];
    float th  = (x - icw) / iw;
    float t1m = th * (1.f - th);
    float den = delta + (d0 + d1 - 2.f * delta) * t1m;
    float num = ih * (delta * th * th + d0 * t1m);
    float outv = ich + num / den;
    float dnum = delta * delta * (d1 * th * th + 2.f * delta * t1m + d0 * (1.f - th) * (1.f - th));
    float lad = logf(dnum) - 2.f * logf(den);
    bool inside = (x1 >= -tb) && (x1 <= tb);
    float res  = inside ? outv : x1;
    float ladr = inside ? lad  : 0.f;

    ZQ[zo]        = res * mk;  // flipped ch0
    ZQ[zo + Tlen] = x0 * mk;   // flipped ch1
    LadAcc[idx] += ladr * mk;
}

// ---------------------------------------------------------------------------
// z_u/z_1 -> z0 = log(clip(dur - sigmoid(z_u)*m, 1e-5))*m ; logdet terms.
// ---------------------------------------------------------------------------
__global__ void sig_dur_k(const float* __restrict__ ZQ, const float* __restrict__ Dur,
                          const float* __restrict__ mask, float* __restrict__ Z,
                          float* __restrict__ LadQ, float* __restrict__ LadZ, int Tlen) {
    int idx = blockIdx.x * 256 + threadIdx.x;
    int t = idx % Tlen; int b = idx / Tlen;
    size_t zo = (size_t)b * 2 * Tlen + t;
    float mk = mask[idx];
    float zu = ZQ[zo], z1 = ZQ[zo + Tlen];
    float u = (1.f / (1.f + expf(-zu))) * mk;
    float z0 = (Dur[idx] - u) * mk;
    // log_sigmoid(zu) + log_sigmoid(-zu) = -softplus(-zu) - softplus(zu)
    LadQ[idx] += (-softplusf(-zu) - softplusf(zu)) * mk;
    float z0l = logf(fmaxf(z0, 1e-5f)) * mk;
    Z[zo] = z0l;
    Z[zo + Tlen] = z1;
    LadZ[idx] = -z0l;   // logdet_sum starts at -sum(z0)
}

// ---------------------------------------------------------------------------
// Final per-batch reduction: out[b] = (nll_b + logs_q_b) / sum_b(len_b)
// ---------------------------------------------------------------------------
__global__ __launch_bounds__(256)
void final_k(const float* __restrict__ EQ, const float* __restrict__ Z,
             const float* __restrict__ LadQ, const float* __restrict__ LadZ,
             const float* __restrict__ mask, const int* __restrict__ lens,
             const float* __restrict__ qls, const float* __restrict__ zls,
             float* __restrict__ out, int Tlen, int Bn) {
    int b = blockIdx.x, tid = threadIdx.x;
    float se = 0.f, sz = 0.f, slq = 0.f, slz = 0.f;
    for (int t = tid; t < Tlen; t += 256) {
        size_t mo = (size_t)b * Tlen + t;
        size_t zo = (size_t)b * 2 * Tlen + t;
        float mk = mask[mo];
        float e0 = EQ[zo], e1 = EQ[zo + Tlen];
        se += (e0 * e0 + e1 * e1) * mk;
        float z0 = Z[zo], z1 = Z[zo + Tlen];
        sz += (z0 * z0 + z1 * z1) * mk;
        slq += LadQ[mo];
        slz += LadZ[mo];
    }
    __shared__ float red[4][256];
    red[0][tid] = se; red[1][tid] = sz; red[2][tid] = slq; red[3][tid] = slz;
    __syncthreads();
    for (int s = 128; s > 0; s >>= 1) {
        if (tid < s) {
            red[0][tid] += red[0][tid + s];
            red[1][tid] += red[1][tid + s];
            red[2][tid] += red[2][tid + s];
            red[3][tid] += red[3][tid + s];
        }
        __syncthreads();
    }
    if (tid == 0) {
        float lenb = (float)lens[b];
        float total = 0.f;
        for (int i = 0; i < Bn; ++i) total += (float)lens[i];
        float logdet_q  = (qls[0] + qls[1]) * lenb + red[2][0];
        float logs_q    = -0.5f * (LOG2PI * 2.f * lenb + red[0][0]) - logdet_q;
        float logdet_sm = (zls[0] + zls[1]) * lenb + red[3][0];
        float nll       = 0.5f * (LOG2PI * 2.f * lenb + red[1][0]) - logdet_sm;
        out[b] = (nll + logs_q) / total;
    }
}

// ---------------------------------------------------------------------------
// Host launch
// ---------------------------------------------------------------------------
extern "C" void kernel_launch(void* const* d_in, const int* in_sizes, int n_in,
                              void* d_out, int out_size, void* d_ws, size_t ws_size,
                              hipStream_t stream) {
    (void)in_sizes; (void)n_in; (void)out_size; (void)ws_size;
    const int Bb = BATCH, Tt = TLEN, Cc = CCALC;

    const float* encodings = (const float*)d_in[0];
    const int*   lengths   = (const int*)d_in[1];
    const float* durations = (const float*)d_in[2];
    const float* e_q       = (const float*)d_in[3];

    int cur = 4;
    auto nf = [&]() { return (const float*)d_in[cur++]; };
    struct ConvP { const float* w; const float* b; };
    struct LnP   { const float* g; const float* b; };
    struct BlockP { ConvP dw; LnP ln1; ConvP pw; LnP ln2; };
    struct FlowP  { ConvP pre; BlockP blk[3]; ConvP proj; };
    auto rdConv = [&]() { ConvP p; p.w = nf(); p.b = nf(); return p; };
    auto rdLn   = [&]() { LnP p; p.g = nf(); p.b = nf(); return p; };
    auto rdBlk  = [&]() { BlockP p; p.dw = rdConv(); p.ln1 = rdLn(); p.pw = rdConv(); p.ln2 = rdLn(); return p; };
    auto rdFlow = [&]() { FlowP f; f.pre = rdConv(); for (int i = 0; i < 3; ++i) f.blk[i] = rdBlk(); f.proj = rdConv(); return f; };

    ConvP prenet_pre = rdConv();
    BlockP prenet_ddsc[3];  for (int i = 0; i < 3; ++i) prenet_ddsc[i] = rdBlk();
    ConvP prenet_post = rdConv();
    ConvP postnet_pre = rdConv();
    BlockP postnet_ddsc[3]; for (int i = 0; i < 3; ++i) postnet_ddsc[i] = rdBlk();
    ConvP postnet_post = rdConv();
    const float* qaff_m = nf(); const float* qaff_ls = nf();
    FlowP qflows[4]; for (int i = 0; i < 4; ++i) qflows[i] = rdFlow();
    const float* aff_m = nf(); const float* aff_ls = nf();
    FlowP zflows[4]; for (int i = 0; i < 4; ++i) zflows[i] = rdFlow();

    // workspace layout
    char* base = (char*)d_ws;
    size_t off = 0;
    auto alloc = [&](size_t bytes) -> void* {
        void* p = base + off;
        off += bytes; off = (off + 255) & ~(size_t)255;
        return p;
    };
    const size_t big = (size_t)Bb * Cc * Tt * sizeof(float);
    float* MASK  = (float*)alloc((size_t)Bb * Tt * 4);
    float* ACT   = (float*)alloc(big);
    float* T1    = (float*)alloc(big);
    float* T2    = (float*)alloc(big);
    float* Xc    = (float*)alloc(big);
    float* CONDQ = (float*)alloc(big);
    float* PROJ  = (float*)alloc((size_t)Bb * 32 * Tt * 4);
    float* ZQ    = (float*)alloc((size_t)Bb * 2 * Tt * 4);
    float* Zb    = (float*)alloc((size_t)Bb * 2 * Tt * 4);
    float* LADQ  = (float*)alloc((size_t)Bb * Tt * 4);
    float* LADZ  = (float*)alloc((size_t)Bb * Tt * 4);
    unsigned short* WPACK = (unsigned short*)alloc((size_t)8 << 20);

    // weight packing
    unsigned short* pk_cur = WPACK;
    auto pack = [&](const float* Wf, int M_real, int K) -> unsigned short* {
        int Mtiles = (M_real + 15) / 16;
        size_t tot = (size_t)Mtiles * (K / 32) * 512;
        pack_weights_k<<<dim3((unsigned)((tot + 255) / 256)), dim3(256), 0, stream>>>(
            Wf, pk_cur, M_real, K, Mtiles);
        unsigned short* r = pk_cur; pk_cur += tot; return r;
    };
    unsigned short* P_pre      = pack(prenet_pre.w, 256, CIN);
    unsigned short* P_prepw[3];  for (int i = 0; i < 3; ++i) P_prepw[i]  = pack(prenet_ddsc[i].pw.w, 256, 256);
    unsigned short* P_prepost  = pack(prenet_post.w, 256, 256);
    unsigned short* P_postpw[3]; for (int i = 0; i < 3; ++i) P_postpw[i] = pack(postnet_ddsc[i].pw.w, 256, 256);
    unsigned short* P_postpost = pack(postnet_post.w, 256, 256);
    unsigned short* P_qpw[4][3]; unsigned short* P_qproj[4];
    for (int f = 0; f < 4; ++f) {
        for (int i = 0; i < 3; ++i) P_qpw[f][i] = pack(qflows[f].blk[i].pw.w, 256, 256);
        P_qproj[f] = pack(qflows[f].proj.w, 29, 256);
    }
    unsigned short* P_zpw[4][3]; unsigned short* P_zproj[4];
    for (int f = 0; f < 4; ++f) {
        for (int i = 0; i < 3; ++i) P_zpw[f][i] = pack(zflows[f].blk[i].pw.w, 256, 256);
        P_zproj[f] = pack(zflows[f].proj.w, 29, 256);
    }

    dim3 g64(Tt / 64, Bb);
    auto gemm = [&](const float* Xin, float* Y, const unsigned short* Ap, const float* bias,
                    const float* mk, int K, int Mtiles, int M_real) {
        gemm_wmma_k<<<g64, 256, 0, stream>>>(Xin, Y, Ap, bias, mk, K, Mtiles, M_real, Tt);
    };
    auto lngelu = [&](const float* Xin, float* Y, LnP ln, const float* res) {
        ln_gelu_k<<<g64, 256, 0, stream>>>(Xin, Y, ln.g, ln.b, res, Tt, Cc);
    };
    auto ddscrun = [&](BlockP* blks, unsigned short** pw) {
        int dil = 1;
        for (int i = 0; i < 3; ++i) {
            dwconv_k<<<g64, 256, 0, stream>>>(ACT, T1, blks[i].dw.w, blks[i].dw.b, MASK, dil, Tt, Cc);
            lngelu(T1, T1, blks[i].ln1, nullptr);
            gemm(T1, T2, pw[i], blks[i].pw.b, nullptr, 256, 16, 256);
            lngelu(T2, ACT, blks[i].ln2, ACT);
            dil *= 3;
        }
    };
    auto flowrun = [&](FlowP& f, unsigned short** pw, unsigned short* pj,
                       float* zbuf, const float* cond, float* lad) {
        ch_expand_k<<<g64, 256, 0, stream>>>(zbuf, 2 * Tt, f.pre.w, f.pre.b, cond, ACT, Tt, Cc);
        ddscrun(f.blk, pw);
        gemm(ACT, PROJ, pj, f.proj.b, MASK, 256, 2, 29);
        rqs_k<<<dim3((Bb * Tt) / 256), 256, 0, stream>>>(PROJ, zbuf, MASK, lad, Tt);
    };

    // pipeline
    mask_k<<<dim3(Tt / 256, Bb), 256, 0, stream>>>(lengths, MASK, Tt);
    fill0_k<<<dim3((Bb * Tt) / 256), 256, 0, stream>>>(LADQ, Bb * Tt);

    // x (condition) path
    gemm(encodings, ACT, P_pre, prenet_pre.b, nullptr, CIN, 16, 256);
    ddscrun(prenet_ddsc, P_prepw);
    gemm(ACT, T2, P_prepost, prenet_post.b, MASK, 256, 16, 256);
    axpy_mask_k<<<dim3((unsigned)(big / 4 / 256)), 256, 0, stream>>>(nullptr, T2, MASK, Xc, Tt, Cc);

    // duration (postnet) path -> cond_q = x + dh
    ch_expand_k<<<g64, 256, 0, stream>>>(durations, Tt, postnet_pre.w, postnet_pre.b, nullptr, ACT, Tt, Cc);
    ddscrun(postnet_ddsc, P_postpw);
    gemm(ACT, T2, P_postpost, postnet_post.b, MASK, 256, 16, 256);
    axpy_mask_k<<<dim3((unsigned)(big / 4 / 256)), 256, 0, stream>>>(Xc, T2, MASK, CONDQ, Tt, Cc);

    // posterior flows
    affine_k<<<dim3((Bb * 2 * Tt) / 256), 256, 0, stream>>>(e_q, qaff_m, qaff_ls, MASK, ZQ, Tt);
    for (int f = 0; f < 4; ++f) {
        unsigned short* pw[3] = {P_qpw[f][0], P_qpw[f][1], P_qpw[f][2]};
        flowrun(qflows[f], pw, P_qproj[f], ZQ, CONDQ, LADQ);
    }
    sig_dur_k<<<dim3((Bb * Tt) / 256), 256, 0, stream>>>(ZQ, durations, MASK, Zb, LADQ, LADZ, Tt);

    // prior flows
    affine_k<<<dim3((Bb * 2 * Tt) / 256), 256, 0, stream>>>(Zb, aff_m, aff_ls, MASK, Zb, Tt);
    for (int f = 0; f < 4; ++f) {
        unsigned short* pw[3] = {P_zpw[f][0], P_zpw[f][1], P_zpw[f][2]};
        flowrun(zflows[f], pw, P_zproj[f], Zb, Xc, LADZ);
    }

    final_k<<<dim3(Bb), 256, 0, stream>>>(e_q, Zb, LADQ, LADZ, MASK, lengths,
                                          qaff_ls, aff_ls, (float*)d_out, Tt, Bb);
}
